// QuantLinearEmbedding_20650202759724
// MI455X (gfx1250) — 8-bit matrix operand order repaired and checked by reading, compiled, not run on hardware
//
#include <hip/hip_runtime.h>
#include <hip/hip_bf16.h>

#define VOCAB   50257
#define EMBED   128
#define NTOK    2048
#define VPAD    50304
#define KCHUNKS 16
#define F32_TINY 1.17549435e-38f

typedef __attribute__((ext_vector_type(8))) int v8i;
typedef __attribute__((ext_vector_type(4))) int v4i;


__global__ void k_init(unsigned* stats, int* acc, int n) {
    int i = blockIdx.x * blockDim.x + threadIdx.x;
    if (i < 2) stats[i] = 0u;
    for (int j = i; j < n; j += gridDim.x * blockDim.x) acc[j] = 0;
}

__global__ void k_absmax(const float* __restrict__ p, long long n, unsigned* out) {
    __shared__ float red[256];
    float m = 0.0f;
    long long stride = (long long)gridDim.x * blockDim.x;
    for (long long i = (long long)blockIdx.x * blockDim.x + threadIdx.x; i < n; i += stride)
        m = fmaxf(m, fabsf(p[i]));
    red[threadIdx.x] = m;
    __syncthreads();
    for (int s = 128; s > 0; s >>= 1) {
        if ((int)threadIdx.x < s) red[threadIdx.x] = fmaxf(red[threadIdx.x], red[threadIdx.x + s]);
        __syncthreads();
    }
    if (threadIdx.x == 0) atomicMax(out, __float_as_uint(red[0]));
}

__global__ void k_quant_w(const float* __restrict__ W, const unsigned* __restrict__ stats,
                          int* __restrict__ Qw_dw) {
    const float sw = fmaxf(__uint_as_float(stats[1]) / 127.0f, F32_TINY);
    const int total = EMBED * (VPAD / 4);
    for (int i = blockIdx.x * blockDim.x + threadIdx.x; i < total; i += gridDim.x * blockDim.x) {
        const int e  = i / (VPAD / 4);
        const int v0 = (i - e * (VPAD / 4)) * 4;
        int packed = 0;
        #pragma unroll
        for (int j = 0; j < 4; ++j) {
            const int v = v0 + j;
            int q = 0;
            if (v < VOCAB) {
                q = __float2int_rn(W[(size_t)e * VOCAB + v] / sw);
                q = q > 127 ? 127 : (q < -127 ? -127 : q);
            }
            packed |= (q & 0xff) << (8 * j);
        }
        Qw_dw[i] = packed;
    }
}

__device__ __forceinline__ int quant4(const float* __restrict__ p, float invs) {
    int r = 0;
    #pragma unroll
    for (int j = 0; j < 4; ++j) {
        int q = __float2int_rn(p[j] * invs);
        q = q > 127 ? 127 : (q < -128 ? -128 : q);
        r |= (q & 0xff) << (8 * j);
    }
    return r;
}

__device__ __forceinline__ int quant4_guard(const float* __restrict__ row, int base, int kend,
                                            float invs) {
    int r = 0;
    #pragma unroll
    for (int j = 0; j < 4; ++j) {
        const int kk = base + j;
        int q = 0;
        if (kk < kend) {
            q = __float2int_rn(row[kk] * invs);
            q = q > 127 ? 127 : (q < -128 ? -128 : q);
        }
        r |= (q & 0xff) << (8 * j);
    }
    return r;
}

__device__ __forceinline__ v8i make_b(const signed char* __restrict__ bp, int k, int hB) {
    v4i blo = *(const v4i*)(bp + k + hB);
    v4i bhi = *(const v4i*)(bp + k + 32 + hB);
    v8i b;
    b[0] = blo[0]; b[1] = blo[1]; b[2] = blo[2]; b[3] = blo[3];
    b[4] = bhi[0]; b[5] = bhi[1]; b[6] = bhi[2]; b[7] = bhi[3];
    return b;
}

__global__ __launch_bounds__(256) void k_gemm(const float* __restrict__ X,
                                              const signed char* __restrict__ Qw,
                                              const unsigned* __restrict__ stats,
                                              int* __restrict__ acc) {
    const int lane = threadIdx.x & 31;
    const int wave = threadIdx.x >> 5;
    const int n0   = blockIdx.x * 128 + wave * 16;

    const int ITER = (VOCAB + 63) >> 6;
    const int CH   = (ITER + KCHUNKS - 1) / KCHUNKS;
    const int kbeg = blockIdx.y * CH * 64;
    const int kend = min(kbeg + CH * 64, VOCAB);

    const float sx    = fmaxf(__uint_as_float(stats[0]) / 127.0f, F32_TINY);
    const float invsx = 1.0f / sx;

    const int hA = (lane & 16) ? 16 : 0;
    const int hB = (lane & 16) ? 16 : 0;
    const float* __restrict__ xrow = X + (size_t)(n0 + (lane & 15)) * VOCAB;
    const signed char* __restrict__ qcol = Qw + (size_t)(lane & 15) * VPAD;

    v8i accv[8];
    #pragma unroll
    for (int t = 0; t < 8; ++t) accv[t] = (v8i){0, 0, 0, 0, 0, 0, 0, 0};

    int k = kbeg;
    for (; k + 64 <= kend; k += 64) {
        v8i a;
        #pragma unroll
        for (int j = 0; j < 4; ++j) {
            const float* p = xrow + k + hA + 8 * j + 16 * (j >> 1);
            a[2 * j + 0] = quant4(p, invsx);
            a[2 * j + 1] = quant4(p + 4, invsx);
        }
        #pragma unroll
        for (int t = 0; t < 8; ++t) {
            v8i b = make_b(qcol + (size_t)t * 16 * VPAD, k, hB);
            accv[t] = __builtin_amdgcn_wmma_i32_16x16x64_iu8(
                 true, a,  true, b, accv[t], false, false);
        }
    }
    if (k < kend) {
        v8i a;
        #pragma unroll
        for (int j = 0; j < 4; ++j) {
            const int base = k + hA + 8 * j + 16 * (j >> 1);
            a[2 * j + 0] = quant4_guard(xrow, base, kend, invsx);
            a[2 * j + 1] = quant4_guard(xrow, base + 4, kend, invsx);
        }
        #pragma unroll
        for (int t = 0; t < 8; ++t) {
            v8i b = make_b(qcol + (size_t)t * 16 * VPAD, k, hB);
            accv[t] = __builtin_amdgcn_wmma_i32_16x16x64_iu8(
                true, a, true, b, accv[t], false, false);
        }
    }

    const int mh  = (lane & 16) ? 8 : 0;
    const int col = lane & 15;
    #pragma unroll
    for (int t = 0; t < 8; ++t) {
        #pragma unroll
        for (int r = 0; r < 8; ++r) {
            const int row = n0 + r + mh;
            atomicAdd(&acc[row * EMBED + (t * 16 + col)], accv[t][r]);
        }
    }
}

__global__ void k_final(const int* __restrict__ acc, const unsigned* __restrict__ stats,
                        float* __restrict__ out, int n) {
    const float sx = fmaxf(__uint_as_float(stats[0]) / 127.0f, F32_TINY);
    const float sw = fmaxf(__uint_as_float(stats[1]) / 127.0f, F32_TINY);
    const float s  = sx * sw;
    for (int i = blockIdx.x * blockDim.x + threadIdx.x; i < n; i += gridDim.x * blockDim.x)
        out[i] = (float)acc[i] * s;
}

extern "C" void kernel_launch(void* const* d_in, const int* in_sizes, int n_in,
                              void* d_out, int out_size, void* d_ws, size_t ws_size,
                              hipStream_t stream) {
    const float* X = (const float*)d_in[0];
    const float* W = (const float*)d_in[1];
    float* out = (float*)d_out;

    unsigned* stats = (unsigned*)d_ws;
    int* acc = (int*)((char*)d_ws + 64);
    signed char* Qw = (signed char*)d_ws + 64 + (size_t)NTOK * EMBED * sizeof(int);

    k_init<<<512, 256, 0, stream>>>(stats, acc, NTOK * EMBED);
    k_absmax<<<2048, 256, 0, stream>>>(X, (long long)NTOK * VOCAB, stats + 0);
    k_absmax<<<256, 256, 0, stream>>>(W, (long long)EMBED * VOCAB, stats + 1);
    k_quant_w<<<1024, 256, 0, stream>>>(W, stats, (int*)Qw);
    dim3 grid(NTOK / 128, KCHUNKS);
    k_gemm<<<grid, 256, 0, stream>>>(X, Qw, stats, acc);
    k_final<<<512, 256, 0, stream>>>(acc, stats, out, NTOK * EMBED);
}
